// SFM_17008070492740
// MI455X (gfx1250) — matrix operand order repaired and checked by reading, not compiled, not run on hardware
//
#include <hip/hip_runtime.h>
#include <hip/hip_bf16.h>

// ---------------------------------------------------------------------------
// CDNA5 (gfx1250, wave32) implementation of the watermark cross-attention block.
// All GEMMs use v_wmma_f32_16x16x32_f16. Fragment layouts follow CDNA5 ISA
// 7.12.2:
//   A (16x32 f16): lane l -> row M=l%16; halves at K = base+0..7 and
//                  16+base+0..7, base = 8*(l>=16)  -> two 16B loads/lane.
//   B (32x16 f16): lane l -> col N=l%16; halves at K = kb+0..7 and 16+kb+0..7,
//                  kb = 8*(l>=16)                  -> two 16B loads/lane.
//   C/D (16x16 f32): vgpr r, lane l -> (M = r + 8*(l>=16), N = l%16).
// Big GEMMs use 64x64 wave tiles (16 WMMA / k-step) to maximize fragment
// reuse and cut L2 re-read amplification 4x vs a 16x64 tile.
// ---------------------------------------------------------------------------

typedef __attribute__((ext_vector_type(16))) _Float16 v16h;
typedef __attribute__((ext_vector_type(8)))  _Float16 v8h;
typedef __attribute__((ext_vector_type(2)))  _Float16 v2h;
typedef __attribute__((ext_vector_type(8)))  float    v8f;

static constexpr int kB  = 16;    // batch
static constexpr int kC  = 512;   // channels
static constexpr int kHW = 1024;  // 32*32 spatial
static constexpr int kNH = 8;     // heads
static constexpr int kHC = 64;    // head channels
static constexpr int kC3 = 1536;  // 3*C
static constexpr int kC2 = 1024;  // 2*C

__device__ __forceinline__ v8f wmma_f16(v16h a, v16h b, v8f c) {
  return __builtin_amdgcn_wmma_f32_16x16x32_f16(false, a, false, b,
                                                (short)0, c, false, false);
}

union U16h { v16h v; v8h h[2]; };

// A fragment: tile = &src[row0*ld + k0] (row-major, rows = M, contiguous K).
__device__ __forceinline__ v16h ld_frag_a(const _Float16* tile, int ld, int lane) {
  int m  = lane & 15;
  int kb = (lane >> 4) << 3;            // 0 or 8
  const _Float16* r = tile + (size_t)m * ld + kb;
  U16h u;
  u.h[0] = *(const v8h*)(r);
  u.h[1] = *(const v8h*)(r + 16);
  return u.v;
}

// B fragment: src holds B^T, i.e. rows = N, contiguous K. tile = &src[n0*ld + k0].
__device__ __forceinline__ v16h ld_frag_b(const _Float16* tile, int ld, int lane) {
  int n  = lane & 15;
  int kb = (lane >> 4) << 3;            // 0 or 8
  const _Float16* r = tile + (size_t)n * ld + kb;
  U16h u;
  u.h[0] = *(const v8h*)(r);
  u.h[1] = *(const v8h*)(r + 16);
  return u.v;
}

// ---------------------------------------------------------------------------
// Kernel 0: weight prep — pointwise weights fp32->f16 (same layout),
//           proj [2C,C] -> projT f16 [C][2C].
// ---------------------------------------------------------------------------
__global__ void __launch_bounds__(256)
prep_kernel(const float* __restrict__ pw_img, const float* __restrict__ pw_wm,
            const float* __restrict__ proj,
            _Float16* __restrict__ wpw, _Float16* __restrict__ projT) {
  int i = blockIdx.x * 256 + threadIdx.x;
  const int NPW = kC3 * kC;
  if (i < NPW) {
    wpw[i] = (_Float16)pw_img[i];
  } else if (i < 2 * NPW) {
    wpw[i] = (_Float16)pw_wm[i - NPW];
  } else {
    int j = i - 2 * NPW;
    if (j < kC * kC2) {
      int c = j / kC2, d = j % kC2;
      projT[(size_t)c * kC2 + d] = (_Float16)proj[(size_t)d * kC + c];
    }
  }
}

// ---------------------------------------------------------------------------
// Kernel 1: LayerNorm over [C,H,W] per sample + affine; writes transposed
//           f16 xnt[b*2+s][p][c]  (p-major so pointwise B-frags are contiguous).
// One block per (b,s); 1024 threads, thread t owns spatial position p=t.
// ---------------------------------------------------------------------------
__global__ void __launch_bounds__(1024)
ln_kernel(const float* __restrict__ img, const float* __restrict__ wm,
          const float* __restrict__ iw, const float* __restrict__ ib,
          const float* __restrict__ ww, const float* __restrict__ wb,
          _Float16* __restrict__ xnt) {
  int bs = blockIdx.x;
  int s = bs & 1, b = bs >> 1;
  const float* x  = (s ? wm : img) + (size_t)b * kC * kHW;
  const float* w  = s ? ww : iw;
  const float* bb = s ? wb : ib;
  int t = threadIdx.x;

  float sum = 0.f, sq = 0.f;
  for (int c = 0; c < kC; ++c) {
    float v = x[(size_t)c * kHW + t];
    sum += v; sq += v * v;
  }
  __shared__ float rs[1024], rq[1024];
  rs[t] = sum; rq[t] = sq;
  __syncthreads();
  for (int off = 512; off > 0; off >>= 1) {
    if (t < off) { rs[t] += rs[t + off]; rq[t] += rq[t + off]; }
    __syncthreads();
  }
  const float inv_n = 1.0f / (float)(kC * kHW);
  float mu  = rs[0] * inv_n;
  float var = rq[0] * inv_n - mu * mu;
  float rstd = rsqrtf(var + 1e-5f);

  _Float16* o = xnt + (size_t)bs * kHW * kC;
  for (int c = 0; c < kC; ++c) {
    size_t idx = (size_t)c * kHW + t;
    float v = (x[idx] - mu) * rstd * w[idx] + bb[idx];
    o[(size_t)t * kC + c] = (_Float16)v;   // thread streams its own 1KB row
  }
}

// ---------------------------------------------------------------------------
// Kernel 2: pointwise 1x1 conv as GEMM: Y[bs][o][p] = W[s][o][c] * xnt[bs][p][c]
//           M=1536 N=1024 K=512 per (b,s). Wave tile = 64(o) x 64(p),
//           16 WMMA per k-step, A/B fragments each reused 4x.
// ---------------------------------------------------------------------------
__global__ void __launch_bounds__(256)
pw_gemm_kernel(const _Float16* __restrict__ wpw, const _Float16* __restrict__ xnt,
               const float* __restrict__ bias_i, const float* __restrict__ bias_w,
               _Float16* __restrict__ Y) {
  int wid  = blockIdx.x * 8 + (threadIdx.x >> 5);
  int lane = threadIdx.x & 31;
  int bs  = wid / (24 * 16);            // 24 o-blocks x 16 p-groups
  int rem = wid % (24 * 16);
  int ob = rem >> 4, pg = rem & 15;
  int s = bs & 1;

  const _Float16* A  = wpw + (size_t)s * kC3 * kC + (size_t)ob * 64 * kC;
  const _Float16* Bt = xnt + (size_t)bs * kHW * kC + (size_t)pg * 64 * kC;

  v8f acc[4][4] = {};
  for (int k = 0; k < kC; k += 32) {
    v16h a[4];
    for (int i = 0; i < 4; ++i)
      a[i] = ld_frag_a(A + (size_t)i * 16 * kC + k, kC, lane);
    for (int j = 0; j < 4; ++j) {
      v16h bf = ld_frag_b(Bt + (size_t)j * 16 * kC + k, kC, lane);
      for (int i = 0; i < 4; ++i)
        acc[i][j] = wmma_f16(a[i], bf, acc[i][j]);
    }
  }
  const float* bias = s ? bias_w : bias_i;
  int nn = lane & 15, mb = (lane >> 4) << 3;
  _Float16* Yo = Y + (size_t)bs * kC3 * kHW + (size_t)pg * 64 + nn;
  for (int i = 0; i < 4; ++i)
    for (int j = 0; j < 4; ++j)
      for (int r = 0; r < 8; ++r) {
        int o = ob * 64 + i * 16 + mb + r;
        Yo[(size_t)o * kHW + j * 16] = (_Float16)(acc[i][j][r] + bias[o]);
      }
}

// ---------------------------------------------------------------------------
// Kernel 3: grouped 3x3 depthwise (groups=C, 3 ch/group), SAME padding,
//           routes Q/K/V into attention-friendly layouts:
//   Q -> q[b,s,n][hc][p]        (GEMM1 B operand, rows contiguous in p)
//   K -> k[b,n][k128][p]        (k128 = s*64+hc; GEMM1 A operand)
//   V -> vT[b,n][p][k128]       (GEMM2 B operand, rows contiguous in k)
// ---------------------------------------------------------------------------
__global__ void __launch_bounds__(256)
dw_kernel(const _Float16* __restrict__ Y,
          const float* __restrict__ dww_i, const float* __restrict__ dwb_i,
          const float* __restrict__ dww_w, const float* __restrict__ dwb_w,
          _Float16* __restrict__ qbuf, _Float16* __restrict__ kbuf,
          _Float16* __restrict__ vtbuf) {
  int p  = blockIdx.x * 256 + threadIdx.x;
  int o  = blockIdx.y;
  int bs = blockIdx.z;
  int s = bs & 1, b = bs >> 1;
  int h = p >> 5, w0 = p & 31;

  const _Float16* Yb = Y + (size_t)bs * kC3 * kHW;
  const float* Wd = (s ? dww_w : dww_i) + o * 27;
  float acc = (s ? dwb_w : dwb_i)[o];
  int g = o / 3;                          // group; input channels 3g..3g+2
  for (int i = 0; i < 3; ++i) {
    const _Float16* ch = Yb + (size_t)(3 * g + i) * kHW;
    for (int kh = 0; kh < 3; ++kh) {
      int hh = h + kh - 1;
      if ((unsigned)hh >= 32u) continue;
      for (int kw = 0; kw < 3; ++kw) {
        int wq = w0 + kw - 1;
        if ((unsigned)wq >= 32u) continue;
        acc += Wd[i * 9 + kh * 3 + kw] * (float)ch[hh * 32 + wq];
      }
    }
  }
  _Float16 v = (_Float16)acc;
  if (o < 512) {                          // Q
    int n = o >> 6, hc = o & 63;
    qbuf[(((size_t)(b * 2 + s) * kNH + n) * kHC + hc) * kHW + p] = v;
  } else if (o < 1024) {                  // K (channel-concat over streams)
    int c2 = o - 512; int n = c2 >> 6; int k128 = s * 64 + (c2 & 63);
    kbuf[(((size_t)(b * kNH + n)) * 128 + k128) * kHW + p] = v;
  } else {                                // V, transposed
    int c2 = o - 1024; int n = c2 >> 6; int k128 = s * 64 + (c2 & 63);
    vtbuf[((size_t)(b * kNH + n) * kHW + p) * 128 + k128] = v;
  }
}

// ---------------------------------------------------------------------------
// Kernel 4: fused attention per (b, n, stream):
//   S[128][64] = Kc @ Q^T (K=1024)  -> LDS
//   row softmax over q (64), written transposed to AT[q][k] f16 in LDS
//   out[q=64][p=1024] = AT @ Vc (K=128), stored to attnT[b][p][d], d=n*128+s*64+q
// 512 threads = 16 waves.
// ---------------------------------------------------------------------------
__global__ void __launch_bounds__(512)
attn_kernel(const _Float16* __restrict__ kbuf, const _Float16* __restrict__ qbuf,
            const _Float16* __restrict__ vtbuf, _Float16* __restrict__ attnT) {
  int s = blockIdx.x, n = blockIdx.y, b = blockIdx.z;
  int t = threadIdx.x, warp = t >> 5, lane = t & 31;

  __shared__ float S[128][68];                          // padded vs bank conflicts
  __shared__ __align__(16) _Float16 AT[64][136];        // softmax^T, padded

  const _Float16* Kc = kbuf  + (size_t)(b * kNH + n) * 128 * kHW;
  const _Float16* Qh = qbuf  + (size_t)((b * 2 + s) * kNH + n) * kHC * kHW;
  const _Float16* Vt = vtbuf + (size_t)(b * kNH + n) * kHW * 128;

  // ---- GEMM1: wave (kt, qts..qts+1) computes two 16x16 score tiles ----
  int kt = warp & 7, qts = (warp >> 3) * 2;
  v8f s0 = {}, s1 = {};
  for (int k = 0; k < kHW; k += 32) {
    v16h a = ld_frag_a(Kc + (size_t)kt * 16 * kHW + k, kHW, lane);
    s0 = wmma_f16(a, ld_frag_b(Qh + (size_t)qts * 16 * kHW + k, kHW, lane), s0);
    s1 = wmma_f16(a, ld_frag_b(Qh + (size_t)(qts + 1) * 16 * kHW + k, kHW, lane), s1);
  }
  int nn = lane & 15, mb = (lane >> 4) << 3;
  for (int r = 0; r < 8; ++r) {
    S[kt * 16 + mb + r][qts * 16 + nn]       = s0[r];
    S[kt * 16 + mb + r][(qts + 1) * 16 + nn] = s1[r];
  }
  __syncthreads();

  // ---- softmax over q per k-row, write transposed f16 ----
  if (t < 128) {
    float m = -3.0e38f;
    for (int q = 0; q < 64; ++q) m = fmaxf(m, S[t][q]);
    float sum = 0.f;
    for (int q = 0; q < 64; ++q) sum += __expf(S[t][q] - m);
    float inv = 1.0f / sum;
    for (int q = 0; q < 64; ++q)
      AT[q][t] = (_Float16)(__expf(S[t][q] - m) * inv);
  }
  __syncthreads();

  // ---- GEMM2: wave (qt, pg2) does 16 p-tiles; A-frags from LDS, reused ----
  int qt = warp & 3, pg2 = warp >> 2;
  v16h af[4];
  for (int ks = 0; ks < 4; ++ks)
    af[ks] = ld_frag_a(&AT[qt * 16][ks * 32], 136, lane);

  _Float16* Ao = attnT + (size_t)b * kHW * kC2
                 + n * 128 + s * 64 + qt * 16 + mb;      // d-offset for this lane
  for (int j = 0; j < 16; ++j) {
    int pt = pg2 * 16 + j;
    v8f acc = {};
    for (int ks = 0; ks < 4; ++ks)
      acc = wmma_f16(af[ks],
                     ld_frag_b(Vt + (size_t)pt * 16 * 128 + ks * 32, 128, lane),
                     acc);
    int p = pt * 16 + nn;
    _Float16* row = Ao + (size_t)p * kC2;
    for (int jj = 0; jj < 4; ++jj) {
      v2h pk = {(_Float16)acc[2 * jj], (_Float16)acc[2 * jj + 1]};
      *(v2h*)(row + 2 * jj) = pk;                        // packed b32 stores
    }
  }
}

// ---------------------------------------------------------------------------
// Kernel 5: final projection: out[b][c][p] = sum_d projT[c][d] * attnT[b][p][d]
//           M=512 N=1024 K=1024 per batch; wave tile 64(c) x 64(p); f32 out.
// ---------------------------------------------------------------------------
__global__ void __launch_bounds__(256)
out_gemm_kernel(const _Float16* __restrict__ projT, const _Float16* __restrict__ attnT,
                float* __restrict__ out) {
  int wid  = blockIdx.x * 8 + (threadIdx.x >> 5);
  int lane = threadIdx.x & 31;
  int b   = wid / (8 * 16);             // 8 c-blocks x 16 p-groups
  int rem = wid % (8 * 16);
  int cb = rem >> 4, pg = rem & 15;

  const _Float16* A  = projT + (size_t)cb * 64 * kC2;
  const _Float16* Bt = attnT + (size_t)b * kHW * kC2 + (size_t)pg * 64 * kC2;

  v8f acc[4][4] = {};
  for (int k = 0; k < kC2; k += 32) {
    v16h a[4];
    for (int i = 0; i < 4; ++i)
      a[i] = ld_frag_a(A + (size_t)i * 16 * kC2 + k, kC2, lane);
    for (int j = 0; j < 4; ++j) {
      v16h bf = ld_frag_b(Bt + (size_t)j * 16 * kC2 + k, kC2, lane);
      for (int i = 0; i < 4; ++i)
        acc[i][j] = wmma_f16(a[i], bf, acc[i][j]);
    }
  }
  int nn = lane & 15, mb = (lane >> 4) << 3;
  float* Ob = out + (size_t)b * kC * kHW + pg * 64 + nn;
  for (int i = 0; i < 4; ++i)
    for (int j = 0; j < 4; ++j)
      for (int r = 0; r < 8; ++r)
        Ob[(size_t)(cb * 64 + i * 16 + mb + r) * kHW + j * 16] = acc[i][j][r];
}

// ---------------------------------------------------------------------------
extern "C" void kernel_launch(void* const* d_in, const int* in_sizes, int n_in,
                              void* d_out, int out_size, void* d_ws, size_t ws_size,
                              hipStream_t stream) {
  (void)in_sizes; (void)n_in; (void)out_size; (void)ws_size;
  const float* image    = (const float*)d_in[0];
  const float* wmark    = (const float*)d_in[1];
  const float* img_ln_w = (const float*)d_in[2];
  const float* img_ln_b = (const float*)d_in[3];
  const float* wm_ln_w  = (const float*)d_in[4];
  const float* wm_ln_b  = (const float*)d_in[5];
  const float* img_pw_w = (const float*)d_in[6];
  const float* img_pw_b = (const float*)d_in[7];
  const float* img_dw_w = (const float*)d_in[8];
  const float* img_dw_b = (const float*)d_in[9];
  const float* wm_pw_w  = (const float*)d_in[10];
  const float* wm_pw_b  = (const float*)d_in[11];
  const float* wm_dw_w  = (const float*)d_in[12];
  const float* wm_dw_b  = (const float*)d_in[13];
  const float* proj     = (const float*)d_in[14];

  // Workspace carve-out (~273 MB), 256B aligned slabs.
  char* ws = (char*)d_ws;
  size_t off = 0;
  auto carve = [&](size_t bytes) -> void* {
    void* p = ws + off;
    off = (off + bytes + 255) & ~(size_t)255;
    return p;
  };
  _Float16* wpw   = (_Float16*)carve((size_t)2 * kC3 * kC * 2);          //  3 MB
  _Float16* projT = (_Float16*)carve((size_t)kC * kC2 * 2);              //  1 MB
  _Float16* xnt   = (_Float16*)carve((size_t)kB * 2 * kHW * kC * 2);     // 32 MB
  _Float16* Ybuf  = (_Float16*)carve((size_t)kB * 2 * kC3 * kHW * 2);    // 96 MB
  _Float16* qbuf  = (_Float16*)carve((size_t)kB * 2 * kNH * kHC * kHW * 2); // 32 MB
  _Float16* kbuf  = (_Float16*)carve((size_t)kB * kNH * 128 * kHW * 2);  // 32 MB
  _Float16* vtbuf = (_Float16*)carve((size_t)kB * kNH * kHW * 128 * 2);  // 32 MB
  _Float16* attnT = (_Float16*)carve((size_t)kB * kHW * kC2 * 2);        // 32 MB

  int prep_n = 2 * kC3 * kC + kC * kC2;
  prep_kernel<<<(prep_n + 255) / 256, 256, 0, stream>>>(
      img_pw_w, wm_pw_w, proj, wpw, projT);

  ln_kernel<<<kB * 2, 1024, 0, stream>>>(
      image, wmark, img_ln_w, img_ln_b, wm_ln_w, wm_ln_b, xnt);

  // 32 (b,s) * 24 o-blocks * 16 p-groups = 12288 waves / 8 per block
  pw_gemm_kernel<<<1536, 256, 0, stream>>>(
      wpw, xnt, img_pw_b, wm_pw_b, Ybuf);

  dw_kernel<<<dim3(4, kC3, kB * 2), 256, 0, stream>>>(
      Ybuf, img_dw_w, img_dw_b, wm_dw_w, wm_dw_b, qbuf, kbuf, vtbuf);

  attn_kernel<<<dim3(2, kNH, kB), 512, 0, stream>>>(
      kbuf, qbuf, vtbuf, attnT);

  // 16 b * 8 c-blocks * 16 p-groups = 2048 waves / 8 per block
  out_gemm_kernel<<<256, 256, 0, stream>>>(
      projT, attnT, (float*)d_out);
}
